// Model_81466939670870
// MI455X (gfx1250) — matrix operand order repaired and checked by reading, not compiled, not run on hardware
//
#include <hip/hip_runtime.h>
#include <hip/hip_bf16.h>
#include <math.h>

typedef __attribute__((ext_vector_type(16))) __bf16 v16bf;
typedef __attribute__((ext_vector_type(8)))  __bf16 v8bf;
typedef __attribute__((ext_vector_type(8)))  float  v8f;

#define NNODE 4096
#define DDIM  128

__device__ __forceinline__ v16bf cat8(v8bf lo, v8bf hi) {
  return __builtin_shufflevector(lo, hi, 0,1,2,3,4,5,6,7,8,9,10,11,12,13,14,15);
}

// ---------------------------------------------------------------------------
// GEMM: C_part[s] = A[M,K] (f32, row-major) x B[K,128] (f32, row-major)
// bf16 WMMA with f32 accumulation. Grid = (M/128, splitK). Each block writes
// its partial 128x128 tile to part + blockIdx.y*M*128 (no atomics -> bitwise
// deterministic). kChunk must be a multiple of 32.
// ---------------------------------------------------------------------------
__global__ __launch_bounds__(256) void gemm_n128(
    const float* __restrict__ A, const float* __restrict__ B,
    float* __restrict__ part, int K, int kChunk) {
  __shared__ __bf16 As[128][40];   // 128 rows x 32 k (pad to 40 for banks)
  __shared__ __bf16 Bt[128][40];   // transposed: 128 n  x 32 k

  const int tid   = threadIdx.x;
  const int M     = gridDim.x * 128;
  const int mBase = blockIdx.x * 128;
  const int k0    = blockIdx.y * kChunk;
  const int kEnd  = k0 + kChunk;

  const int lane = tid & 31;
  const int wave = tid >> 5;
  const int wm   = wave >> 1;      // 0..3 -> 32-row strip
  const int wn   = wave & 1;       // 0..1 -> 64-col strip
  const int l16  = lane & 15;
  const int lh   = lane >> 4;      // lane half

  v8f acc[2][4];
  #pragma unroll
  for (int a = 0; a < 2; ++a)
    #pragma unroll
    for (int b = 0; b < 4; ++b)
      #pragma unroll
      for (int q = 0; q < 8; ++q) acc[a][b][q] = 0.f;

  const int arow = tid >> 1, akg = (tid & 1) * 16;   // A loader mapping
  const int bkk  = tid >> 3, bng = (tid & 7) * 16;   // B loader mapping

  for (int ks = k0; ks < kEnd; ks += 32) {
    // stage A tile (f32 -> bf16), coalesced 16B global loads
    const float4* ap = (const float4*)(A + (size_t)(mBase + arow) * K + ks + akg);
    #pragma unroll
    for (int j = 0; j < 4; ++j) {
      float4 v = ap[j];
      As[arow][akg + 4*j + 0] = (__bf16)v.x;
      As[arow][akg + 4*j + 1] = (__bf16)v.y;
      As[arow][akg + 4*j + 2] = (__bf16)v.z;
      As[arow][akg + 4*j + 3] = (__bf16)v.w;
    }
    // stage B tile transposed (coalesced global reads, scattered LDS writes)
    const float4* bp = (const float4*)(B + (size_t)(ks + bkk) * 128 + bng);
    #pragma unroll
    for (int j = 0; j < 4; ++j) {
      float4 v = bp[j];
      Bt[bng + 4*j + 0][bkk] = (__bf16)v.x;
      Bt[bng + 4*j + 1][bkk] = (__bf16)v.y;
      Bt[bng + 4*j + 2][bkk] = (__bf16)v.z;
      Bt[bng + 4*j + 3][bkk] = (__bf16)v.w;
    }
    __syncthreads();

    // A frag (ISA 16-bit A 16x32): lane l16 = M row, lh selects k-octet.
    v16bf afrag[2], bfrag[4];
    #pragma unroll
    for (int mi = 0; mi < 2; ++mi) {
      int m  = wm*32 + mi*16 + l16;
      int kb = lh*8;
      v8bf lo = *(const v8bf*)&As[m][kb];
      v8bf hi = *(const v8bf*)&As[m][kb + 16];
      afrag[mi] = cat8(lo, hi);
    }
    // B frag: lane l16 = N col, lh selects k-octet (same k order as A).
    #pragma unroll
    for (int ni = 0; ni < 4; ++ni) {
      int n  = wn*64 + ni*16 + l16;
      int kb = lh*8;
      v8bf lo = *(const v8bf*)&Bt[n][kb];
      v8bf hi = *(const v8bf*)&Bt[n][kb + 16];
      bfrag[ni] = cat8(lo, hi);
    }

    #pragma unroll
    for (int mi = 0; mi < 2; ++mi)
      #pragma unroll
      for (int ni = 0; ni < 4; ++ni)
        acc[mi][ni] = __builtin_amdgcn_wmma_f32_16x16x32_bf16(
            false, afrag[mi], false, bfrag[ni], (short)0, acc[mi][ni],
            false, false);
    __syncthreads();
  }

  // write partial tile (C layout: vgpr r -> row lh*8+r, lane l16 -> col)
  float* Cp = part + (size_t)blockIdx.y * M * 128;
  #pragma unroll
  for (int mi = 0; mi < 2; ++mi)
    #pragma unroll
    for (int ni = 0; ni < 4; ++ni) {
      int col = wn*64 + ni*16 + l16;
      #pragma unroll
      for (int r = 0; r < 8; ++r) {
        int row = mBase + wm*32 + mi*16 + lh*8 + r;
        Cp[(size_t)row * 128 + col] = acc[mi][ni][r];
      }
    }
}

// ---------------------------------------------------------------------------
// Split-K reduction + optional bias (per column, N=128) + optional ReLU.
// ---------------------------------------------------------------------------
__global__ void reduce_epi(const float* __restrict__ part, int S, int n,
                           const float* __restrict__ bias, int do_relu,
                           float* __restrict__ out) {
  int i = blockIdx.x * blockDim.x + threadIdx.x;
  if (i >= n) return;
  float a = bias ? bias[i & 127] : 0.f;
  for (int s = 0; s < S; ++s) a += part[(size_t)s * n + i];
  if (do_relu) a = fmaxf(a, 0.f);
  out[i] = a;
}

// ---------------------------------------------------------------------------
// Row sums of a 4096-wide f32 matrix (one block per row).
// ---------------------------------------------------------------------------
__global__ __launch_bounds__(256) void rowsum4096(const float* __restrict__ R,
                                                  float* __restrict__ s) {
  __shared__ float red[256];
  const int row = blockIdx.x;
  const float4* rp = (const float4*)(R + (size_t)row * 4096);
  float acc = 0.f;
  #pragma unroll
  for (int j = 0; j < 4; ++j) {
    float4 v = rp[threadIdx.x + 256 * j];
    acc += v.x + v.y + v.z + v.w;
  }
  red[threadIdx.x] = acc;
  __syncthreads();
  for (int off = 128; off > 0; off >>= 1) {
    if (threadIdx.x < off) red[threadIdx.x] += red[threadIdx.x + off];
    __syncthreads();
  }
  if (threadIdx.x == 0) s[row] = red[0];
}

// Column sums of [4096,128] f32 weight matrix (one block per column).
__global__ __launch_bounds__(256) void colsum128(const float* __restrict__ W,
                                                 float* __restrict__ c) {
  __shared__ float red[256];
  const int col = blockIdx.x;
  float acc = 0.f;
  for (int r = threadIdx.x; r < 4096; r += 256)
    acc += W[(size_t)r * 128 + col];
  red[threadIdx.x] = acc;
  __syncthreads();
  for (int off = 128; off > 0; off >>= 1) {
    if (threadIdx.x < off) red[threadIdx.x] += red[threadIdx.x + off];
    __syncthreads();
  }
  if (threadIdx.x == 0) c[col] = red[0];
}

// ---------------------------------------------------------------------------
// Gather + attention mix. vX[n,d] = relu(rowsum(review)[n] * colsum(Wr)[d]).
// ---------------------------------------------------------------------------
__global__ __launch_bounds__(128) void gather_mix(
    const float* __restrict__ RA, const float* __restrict__ RB,
    const float* __restrict__ sA, const float* __restrict__ sB,
    const float* __restrict__ cA, const float* __restrict__ cB,
    const float* __restrict__ attA, const float* __restrict__ attB,
    const int* __restrict__ u, const int* __restrict__ iidx,
    const int* __restrict__ domain,
    float* __restrict__ Uin, float* __restrict__ Iin) {
  const int b  = blockIdx.x;
  const int dd = threadIdx.x;
  const int uu = u[b];
  const int ii = iidx[b];
  const int dom = domain[0];

  float vA = sA[uu] * cA[dd]; vA = vA > 0.f ? vA : 0.f;
  float vB = sB[uu] * cB[dd]; vB = vB > 0.f ? vB : 0.f;
  float uA = RA[(size_t)uu * 128 + dd] + vA;
  float uB = RB[(size_t)uu * 128 + dd] + vB;

  float user, item;
  if (dom == 0) {
    float w = attA[(size_t)uu * 128 + dd];
    user = uA * w + uB * (1.f - w);
    item = RA[(size_t)ii * 128 + dd];
  } else {
    float w = attB[(size_t)uu * 128 + dd];
    user = uB * w + uA * (1.f - w);
    item = RB[(size_t)ii * 128 + dd];
  }
  Uin[(size_t)b * 128 + dd] = user;
  Iin[(size_t)b * 128 + dd] = item;
}

// ---------------------------------------------------------------------------
extern "C" void kernel_launch(void* const* d_in, const int* in_sizes, int n_in,
                              void* d_out, int out_size, void* d_ws, size_t ws_size,
                              hipStream_t stream) {
  const float* rating_A = (const float*)d_in[0];
  const float* rating_B = (const float*)d_in[1];
  const float* review_A = (const float*)d_in[2];
  const float* review_B = (const float*)d_in[3];
  const float* Wg_A = (const float*)d_in[4];
  const float* Wg_B = (const float*)d_in[5];
  const float* Wr_A = (const float*)d_in[6];
  const float* Wr_B = (const float*)d_in[7];
  const float* att_A = (const float*)d_in[8];
  const float* att_B = (const float*)d_in[9];
  const float* uW1 = (const float*)d_in[10];
  const float* ub1 = (const float*)d_in[11];
  const float* uW2 = (const float*)d_in[12];
  const float* ub2 = (const float*)d_in[13];
  const float* iW1 = (const float*)d_in[14];
  const float* ib1 = (const float*)d_in[15];
  const float* iW2 = (const float*)d_in[16];
  const float* ib2 = (const float*)d_in[17];
  const int* u_idx  = (const int*)d_in[18];
  const int* i_idx  = (const int*)d_in[19];
  const int* domain = (const int*)d_in[20];

  const size_t ND = (size_t)NNODE * DDIM;   // 524288
  float* ws   = (float*)d_ws;
  float* part = ws;                // 8*ND  (split-K partials, reused)
  float* X1A  = ws + 8 * ND;
  float* X1B  = ws + 9 * ND;
  float* RA   = ws + 10 * ND;
  float* RB   = ws + 11 * ND;
  float* Uin  = ws + 12 * ND;
  float* Iin  = ws + 13 * ND;
  float* H1u  = ws + 14 * ND;
  float* H1i  = ws + 15 * ND;
  float* sA   = ws + 16 * ND;
  float* sB   = sA + NNODE;
  float* cA   = sB + NNODE;
  float* cB   = cA + DDIM;

  const int nND   = (int)ND;
  const int redg  = (nND + 255) / 256;
  const dim3 gBig(NNODE / 128, 8);   // split-K 8, kChunk 512
  const dim3 gMlp(NNODE / 128, 4);   // split-K 4, kChunk 32

  // GCN stage 1: X1 = rating @ Wg
  gemm_n128<<<gBig, 256, 0, stream>>>(rating_A, Wg_A, part, NNODE, 512);
  reduce_epi<<<redg, 256, 0, stream>>>(part, 8, nND, nullptr, 0, X1A);
  gemm_n128<<<gBig, 256, 0, stream>>>(rating_B, Wg_B, part, NNODE, 512);
  reduce_epi<<<redg, 256, 0, stream>>>(part, 8, nND, nullptr, 0, X1B);

  // GCN stage 2: R = relu(rating @ X1)
  gemm_n128<<<gBig, 256, 0, stream>>>(rating_A, X1A, part, NNODE, 512);
  reduce_epi<<<redg, 256, 0, stream>>>(part, 8, nND, nullptr, 1, RA);
  gemm_n128<<<gBig, 256, 0, stream>>>(rating_B, X1B, part, NNODE, 512);
  reduce_epi<<<redg, 256, 0, stream>>>(part, 8, nND, nullptr, 1, RB);

  // review GCN collapses to rank-1: rowsum(review) x colsum(Wr)
  rowsum4096<<<NNODE, 256, 0, stream>>>(review_A, sA);
  rowsum4096<<<NNODE, 256, 0, stream>>>(review_B, sB);
  colsum128<<<DDIM, 256, 0, stream>>>(Wr_A, cA);
  colsum128<<<DDIM, 256, 0, stream>>>(Wr_B, cB);

  // gather + attention mix
  gather_mix<<<NNODE, 128, 0, stream>>>(RA, RB, sA, sB, cA, cB, att_A, att_B,
                                        u_idx, i_idx, domain, Uin, Iin);

  float* out_user = (float*)d_out;
  float* out_item = out_user + ND;

  // user MLP
  gemm_n128<<<gMlp, 256, 0, stream>>>(Uin, uW1, part, DDIM, 32);
  reduce_epi<<<redg, 256, 0, stream>>>(part, 4, nND, ub1, 1, H1u);
  gemm_n128<<<gMlp, 256, 0, stream>>>(H1u, uW2, part, DDIM, 32);
  reduce_epi<<<redg, 256, 0, stream>>>(part, 4, nND, ub2, 0, out_user);

  // item MLP
  gemm_n128<<<gMlp, 256, 0, stream>>>(Iin, iW1, part, DDIM, 32);
  reduce_epi<<<redg, 256, 0, stream>>>(part, 4, nND, ib1, 1, H1i);
  gemm_n128<<<gMlp, 256, 0, stream>>>(H1i, iW2, part, DDIM, 32);
  reduce_epi<<<redg, 256, 0, stream>>>(part, 4, nND, ib2, 0, out_item);
}
